// TripletLoss_77472620085346
// MI455X (gfx1250) — matrix operand order repaired and checked by reading, not compiled, not run on hardware
//
#include <hip/hip_runtime.h>
#include <hip/hip_bf16.h>

typedef __attribute__((ext_vector_type(16))) __bf16 v16bf;
typedef __attribute__((ext_vector_type(8)))  __bf16 v8bf;
typedef __attribute__((ext_vector_type(2)))  __bf16 v2bf;
typedef __attribute__((ext_vector_type(8)))  float  v8f;

#define MARGIN 0.3f
#define EPS 1e-12f
#define DMAX 2048

union V16U { v16bf v; v8bf h[2]; };

// ---------------------------------------------------------------------------
// Kernel 1a (fast path): fused bf16x2 split (X -> Xhi + Xlo) + row sq-norms.
// ---------------------------------------------------------------------------
__global__ __launch_bounds__(256) void split_kernel(
    const float* __restrict__ X, __bf16* __restrict__ Xhi,
    __bf16* __restrict__ Xlo, float* __restrict__ sq, int d) {
    const int row = blockIdx.x;
    const float* xr = X + (size_t)row * d;
    float s = 0.f;
    for (int k = threadIdx.x * 2; k < d; k += 512) {
        float x0 = xr[k], x1 = xr[k + 1];
        s += x0 * x0 + x1 * x1;
        __bf16 h0 = (__bf16)x0, h1 = (__bf16)x1;
        v2bf hp; hp[0] = h0; hp[1] = h1;
        v2bf lp; lp[0] = (__bf16)(x0 - (float)h0); lp[1] = (__bf16)(x1 - (float)h1);
        *(v2bf*)(Xhi + (size_t)row * d + k) = hp;
        *(v2bf*)(Xlo + (size_t)row * d + k) = lp;
    }
    __shared__ float buf[256];
    buf[threadIdx.x] = s;
    __syncthreads();
    for (int off = 128; off > 0; off >>= 1) {
        if (threadIdx.x < off) buf[threadIdx.x] += buf[threadIdx.x + off];
        __syncthreads();
    }
    if (threadIdx.x == 0) sq[row] = buf[0];
}

// ---------------------------------------------------------------------------
// Kernel 1b (fallback path): sq-norms only.
// ---------------------------------------------------------------------------
__global__ __launch_bounds__(256) void sq_kernel(const float* __restrict__ X,
                                                 float* __restrict__ sq, int d) {
    const int row = blockIdx.x;
    float s = 0.f;
    for (int k = threadIdx.x; k < d; k += 256) {
        float v = X[(size_t)row * d + k];
        s += v * v;
    }
    __shared__ float buf[256];
    buf[threadIdx.x] = s;
    __syncthreads();
    for (int off = 128; off > 0; off >>= 1) {
        if (threadIdx.x < off) buf[threadIdx.x] += buf[threadIdx.x + off];
        __syncthreads();
    }
    if (threadIdx.x == 0) sq[row] = buf[0];
}

// ---------------------------------------------------------------------------
// Mining combine (extreme value, then min index: order independent, matches
// JAX argmax/argmin first-occurrence tie-break).
// ---------------------------------------------------------------------------
__device__ __forceinline__ void comb_max(float& d, int& i, float od, int oi) {
    if (od > d || (od == d && oi < i)) { d = od; i = oi; }
}
__device__ __forceinline__ void comb_min(float& d, int& i, float od, int oi) {
    if (od < d || (od == d && oi < i)) { d = od; i = oi; }
}

// ---------------------------------------------------------------------------
// Kernel 2 (fast path): fused Gram + batch-hard mining from pre-split bf16.
// Workgroup = 16 rows; A-panel (hi+lo) staged in LDS (128 KB). Each wave
// computes a 16x32 strip (two 16x16 tiles, two accumulators) so the A
// fragments are reused: per K-step 4x ds_load_b128 + 8x global_load_b128
// feed 6 WMMAs (2.0 loads/WMMA). Mining on clamped squared distances.
// ---------------------------------------------------------------------------
__global__ __launch_bounds__(256) void mine_kernel_pre(
    const __bf16* __restrict__ Xhi, const __bf16* __restrict__ Xlo,
    const float* __restrict__ sq, const long long* __restrict__ tgt,
    int* __restrict__ pidx, int* __restrict__ nidx, int n, int d) {

    __shared__ __bf16 sAhi[16 * DMAX];
    __shared__ __bf16 sAlo[16 * DMAX];
    __shared__ float red_pd[8][16]; __shared__ int red_pi[8][16];
    __shared__ float red_nd[8][16]; __shared__ int red_ni[8][16];

    const int r0   = blockIdx.x * 16;
    const int wave = threadIdx.x >> 5;
    const int lane = threadIdx.x & 31;
    const int half = lane >> 4;
    const int nn   = lane & 15;

    // cooperative stage of the A panel (hi+lo) into LDS, 128-bit chunks
    {
        const uint4* shi = (const uint4*)(Xhi + (size_t)r0 * d);
        const uint4* slo = (const uint4*)(Xlo + (size_t)r0 * d);
        uint4* dhi = (uint4*)sAhi;
        uint4* dlo = (uint4*)sAlo;
        const int nchunk = (16 * d) / 8;
        for (int i = threadIdx.x; i < nchunk; i += 256) {
            dhi[i] = shi[i];
            dlo[i] = slo[i];
        }
    }
    __syncthreads();

    float sqr[8]; int tr[8];
    #pragma unroll
    for (int r = 0; r < 8; ++r) {
        int row = r0 + r + 8 * half;
        sqr[r] = sq[row];
        tr[r]  = (int)tgt[row];
    }

    float bpd[8], bnd[8]; int bpi[8], bni[8];
    #pragma unroll
    for (int r = 0; r < 8; ++r) {
        bpd[r] = -1.0f;   bpi[r] = 0x7fffffff;
        bnd[r] = 3.4e38f; bni[r] = 0x7fffffff;
    }

    const __bf16* arow_hi = sAhi + (size_t)nn * d;   // A: M = lane&15 (LDS)
    const __bf16* arow_lo = sAlo + (size_t)nn * d;

    const int ngrp = n >> 5;                          // 32-column groups
    for (int cg = wave; cg < ngrp; cg += 8) {
        const int cb = cg * 32;
        v8f acc0 = {0.f, 0.f, 0.f, 0.f, 0.f, 0.f, 0.f, 0.f};
        v8f acc1 = {0.f, 0.f, 0.f, 0.f, 0.f, 0.f, 0.f, 0.f};
        const __bf16* b0_hi = Xhi + (size_t)(cb + nn) * d;        // tile 0
        const __bf16* b0_lo = Xlo + (size_t)(cb + nn) * d;
        const __bf16* b1_hi = Xhi + (size_t)(cb + 16 + nn) * d;   // tile 1
        const __bf16* b1_lo = Xlo + (size_t)(cb + 16 + nn) * d;

        #pragma unroll 2
        for (int k0 = 0; k0 < d; k0 += 32) {
            V16U ahi, alo, b0h, b0l, b1h, b1l;
            const int ka0 = k0 + 16 * half;      // A: one contiguous 16-run
            ahi.h[0] = *(const v8bf*)(arow_hi + ka0);
            ahi.h[1] = *(const v8bf*)(arow_hi + ka0 + 8);
            alo.h[0] = *(const v8bf*)(arow_lo + ka0);
            alo.h[1] = *(const v8bf*)(arow_lo + ka0 + 8);
            const int kb0 = k0 + 16 * half;      // B: one contiguous 16-run
            b0h.h[0] = *(const v8bf*)(b0_hi + kb0);
            b0h.h[1] = *(const v8bf*)(b0_hi + kb0 + 8);
            b0l.h[0] = *(const v8bf*)(b0_lo + kb0);
            b0l.h[1] = *(const v8bf*)(b0_lo + kb0 + 8);
            b1h.h[0] = *(const v8bf*)(b1_hi + kb0);
            b1h.h[1] = *(const v8bf*)(b1_hi + kb0 + 8);
            b1l.h[0] = *(const v8bf*)(b1_lo + kb0);
            b1l.h[1] = *(const v8bf*)(b1_lo + kb0 + 8);

            acc0 = __builtin_amdgcn_wmma_f32_16x16x32_bf16(false, ahi.v, false, b0h.v,
                                                           (short)0, acc0, false, false);
            acc0 = __builtin_amdgcn_wmma_f32_16x16x32_bf16(false, ahi.v, false, b0l.v,
                                                           (short)0, acc0, false, false);
            acc0 = __builtin_amdgcn_wmma_f32_16x16x32_bf16(false, alo.v, false, b0h.v,
                                                           (short)0, acc0, false, false);
            acc1 = __builtin_amdgcn_wmma_f32_16x16x32_bf16(false, ahi.v, false, b1h.v,
                                                           (short)0, acc1, false, false);
            acc1 = __builtin_amdgcn_wmma_f32_16x16x32_bf16(false, ahi.v, false, b1l.v,
                                                           (short)0, acc1, false, false);
            acc1 = __builtin_amdgcn_wmma_f32_16x16x32_bf16(false, alo.v, false, b1h.v,
                                                           (short)0, acc1, false, false);
        }

        // mining epilogue on clamped squared distances (sqrt is monotone)
        #pragma unroll
        for (int t = 0; t < 2; ++t) {
            const int   col = cb + 16 * t + nn;
            const float sqc = sq[col];
            const int   tc  = (int)tgt[col];
            const v8f&  acc = t ? acc1 : acc0;
            #pragma unroll
            for (int r = 0; r < 8; ++r) {
                int   row = r0 + r + 8 * half;
                float d2  = fmaxf(sqr[r] + sqc - 2.0f * acc[r], EPS);
                if (tr[r] == tc) {
                    if (row != col) comb_max(bpd[r], bpi[r], d2, col);
                } else {
                    comb_min(bnd[r], bni[r], d2, col);
                }
            }
        }
    }

    #pragma unroll
    for (int r = 0; r < 8; ++r) {
        float pd = bpd[r]; int pi = bpi[r];
        float nd = bnd[r]; int ni = bni[r];
        #pragma unroll
        for (int m = 8; m >= 1; m >>= 1) {
            comb_max(pd, pi, __shfl_xor(pd, m, 32), __shfl_xor(pi, m, 32));
            comb_min(nd, ni, __shfl_xor(nd, m, 32), __shfl_xor(ni, m, 32));
        }
        if (nn == 0) {
            int mrow = r + 8 * half;
            red_pd[wave][mrow] = pd; red_pi[wave][mrow] = pi;
            red_nd[wave][mrow] = nd; red_ni[wave][mrow] = ni;
        }
    }
    __syncthreads();

    if (threadIdx.x < 16) {
        int t = threadIdx.x;
        float pd = red_pd[0][t]; int pi = red_pi[0][t];
        float nd = red_nd[0][t]; int ni = red_ni[0][t];
        for (int w = 1; w < 8; ++w) {
            comb_max(pd, pi, red_pd[w][t], red_pi[w][t]);
            comb_min(nd, ni, red_nd[w][t], red_ni[w][t]);
        }
        pidx[r0 + t] = pi;
        nidx[r0 + t] = ni;
    }
}

// ---------------------------------------------------------------------------
// Kernel 2 (fallback): on-the-fly fp32 -> bf16x2 conversion (no big ws).
// ---------------------------------------------------------------------------
__global__ __launch_bounds__(256) void mine_kernel_fly(
    const float* __restrict__ X, const float* __restrict__ sq,
    const long long* __restrict__ tgt,
    int* __restrict__ pidx, int* __restrict__ nidx, int n, int d) {

    __shared__ float red_pd[8][16]; __shared__ int red_pi[8][16];
    __shared__ float red_nd[8][16]; __shared__ int red_ni[8][16];

    const int r0   = blockIdx.x * 16;
    const int wave = threadIdx.x >> 5;
    const int lane = threadIdx.x & 31;
    const int half = lane >> 4;
    const int nn   = lane & 15;

    float sqr[8]; int tr[8];
    #pragma unroll
    for (int r = 0; r < 8; ++r) {
        int row = r0 + r + 8 * half;
        sqr[r] = sq[row];
        tr[r]  = (int)tgt[row];
    }
    float bpd[8], bnd[8]; int bpi[8], bni[8];
    #pragma unroll
    for (int r = 0; r < 8; ++r) {
        bpd[r] = -1.0f;   bpi[r] = 0x7fffffff;
        bnd[r] = 3.4e38f; bni[r] = 0x7fffffff;
    }

    const int ntiles = n >> 4;
    for (int ct = wave; ct < ntiles; ct += 8) {
        const int cb = ct * 16;
        v8f acc = {0.f, 0.f, 0.f, 0.f, 0.f, 0.f, 0.f, 0.f};
        const float* arow = X + (size_t)(r0 + nn) * d;
        const float* brow = X + (size_t)(cb + nn) * d;

        for (int k0 = 0; k0 < d; k0 += 32) {
            v16bf ahi, alo, bhi, blo;
            #pragma unroll
            for (int v = 0; v < 8; ++v) {
                int ka = k0 + 2 * v + 16 * half;
                float a0 = arow[ka], a1 = arow[ka + 1];
                __bf16 h0 = (__bf16)a0, h1 = (__bf16)a1;
                ahi[2 * v] = h0; ahi[2 * v + 1] = h1;
                alo[2 * v] = (__bf16)(a0 - (float)h0);
                alo[2 * v + 1] = (__bf16)(a1 - (float)h1);
                int kb = k0 + 2 * v + 16 * half;
                float b0 = brow[kb], b1 = brow[kb + 1];
                __bf16 g0 = (__bf16)b0, g1 = (__bf16)b1;
                bhi[2 * v] = g0; bhi[2 * v + 1] = g1;
                blo[2 * v] = (__bf16)(b0 - (float)g0);
                blo[2 * v + 1] = (__bf16)(b1 - (float)g1);
            }
            acc = __builtin_amdgcn_wmma_f32_16x16x32_bf16(false, ahi, false, bhi,
                                                          (short)0, acc, false, false);
            acc = __builtin_amdgcn_wmma_f32_16x16x32_bf16(false, ahi, false, blo,
                                                          (short)0, acc, false, false);
            acc = __builtin_amdgcn_wmma_f32_16x16x32_bf16(false, alo, false, bhi,
                                                          (short)0, acc, false, false);
        }

        const int   col = cb + nn;
        const float sqc = sq[col];
        const int   tc  = (int)tgt[col];
        #pragma unroll
        for (int r = 0; r < 8; ++r) {
            int   row = r0 + r + 8 * half;
            float d2  = fmaxf(sqr[r] + sqc - 2.0f * acc[r], EPS);
            if (tr[r] == tc) {
                if (row != col) comb_max(bpd[r], bpi[r], d2, col);
            } else {
                comb_min(bnd[r], bni[r], d2, col);
            }
        }
    }

    #pragma unroll
    for (int r = 0; r < 8; ++r) {
        float pd = bpd[r]; int pi = bpi[r];
        float nd = bnd[r]; int ni = bni[r];
        #pragma unroll
        for (int m = 8; m >= 1; m >>= 1) {
            comb_max(pd, pi, __shfl_xor(pd, m, 32), __shfl_xor(pi, m, 32));
            comb_min(nd, ni, __shfl_xor(nd, m, 32), __shfl_xor(ni, m, 32));
        }
        if (nn == 0) {
            int mrow = r + 8 * half;
            red_pd[wave][mrow] = pd; red_pi[wave][mrow] = pi;
            red_nd[wave][mrow] = nd; red_ni[wave][mrow] = ni;
        }
    }
    __syncthreads();
    if (threadIdx.x < 16) {
        int t = threadIdx.x;
        float pd = red_pd[0][t]; int pi = red_pi[0][t];
        float nd = red_nd[0][t]; int ni = red_ni[0][t];
        for (int w = 1; w < 8; ++w) {
            comb_max(pd, pi, red_pd[w][t], red_pi[w][t]);
            comb_min(nd, ni, red_nd[w][t], red_ni[w][t]);
        }
        pidx[r0 + t] = pi;
        nidx[r0 + t] = ni;
    }
}

// ---------------------------------------------------------------------------
// Kernel 3: one wave per row. Exact fp32 distances for mined pairs (global
// and local features); per-row hinge terms to workspace.
// ---------------------------------------------------------------------------
__global__ __launch_bounds__(256) void local_kernel(
    const float* __restrict__ X, const float* __restrict__ L,
    const float* __restrict__ sq,
    const int* __restrict__ pidx, const int* __restrict__ nidx,
    float* __restrict__ loss_g, float* __restrict__ loss_l,
    int n, int d, int dl) {

    const int wave = blockIdx.x * (blockDim.x >> 5) + (threadIdx.x >> 5);
    const int lane = threadIdx.x & 31;
    if (wave >= n) return;

    const int i = wave, p = pidx[i], q = nidx[i];

    const float* xi = X + (size_t)i * d;
    const float* xp = X + (size_t)p * d;
    const float* xq = X + (size_t)q * d;
    float gip = 0.f, gin = 0.f;
    for (int k = lane; k < d; k += 32) {
        float a = xi[k];
        gip += a * xp[k];
        gin += a * xq[k];
    }

    const float* li = L + (size_t)i * dl;
    const float* lp = L + (size_t)p * dl;
    const float* lq = L + (size_t)q * dl;
    float sii = 0.f, spp = 0.f, sqq = 0.f, sip = 0.f, siq = 0.f;
    for (int k = lane; k < dl; k += 32) {
        float a = li[k], b = lp[k], c = lq[k];
        sii += a * a; spp += b * b; sqq += c * c; sip += a * b; siq += a * c;
    }

    #pragma unroll
    for (int m = 16; m >= 1; m >>= 1) {
        gip += __shfl_xor(gip, m, 32);
        gin += __shfl_xor(gin, m, 32);
        sii += __shfl_xor(sii, m, 32);
        spp += __shfl_xor(spp, m, 32);
        sqq += __shfl_xor(sqq, m, 32);
        sip += __shfl_xor(sip, m, 32);
        siq += __shfl_xor(siq, m, 32);
    }

    if (lane == 0) {
        float dap = sqrtf(fmaxf(sq[i] + sq[p] - 2.0f * gip, EPS));
        float dan = sqrtf(fmaxf(sq[i] + sq[q] - 2.0f * gin, EPS));
        float lap = sqrtf(fmaxf(sii + spp - 2.0f * sip, EPS));
        float lan = sqrtf(fmaxf(sii + sqq - 2.0f * siq, EPS));
        loss_g[i] = fmaxf(dap - dan + MARGIN, 0.f);
        loss_l[i] = fmaxf(lap - lan + MARGIN, 0.f);
    }
}

// ---------------------------------------------------------------------------
// Kernel 4: deterministic final mean of both hinge arrays -> d_out[0..1]
// ---------------------------------------------------------------------------
__global__ __launch_bounds__(256) void reduce_kernel(
    const float* __restrict__ loss_g, const float* __restrict__ loss_l,
    float* __restrict__ out, int n) {
    __shared__ float bg[256], bl[256];
    float sg = 0.f, sl = 0.f;
    for (int i = threadIdx.x; i < n; i += 256) { sg += loss_g[i]; sl += loss_l[i]; }
    bg[threadIdx.x] = sg; bl[threadIdx.x] = sl;
    __syncthreads();
    for (int off = 128; off > 0; off >>= 1) {
        if (threadIdx.x < off) {
            bg[threadIdx.x] += bg[threadIdx.x + off];
            bl[threadIdx.x] += bl[threadIdx.x + off];
        }
        __syncthreads();
    }
    if (threadIdx.x == 0) {
        out[0] = bg[0] / (float)n;
        out[1] = bl[0] / (float)n;
    }
}

// ---------------------------------------------------------------------------
extern "C" void kernel_launch(void* const* d_in, const int* in_sizes, int n_in,
                              void* d_out, int out_size, void* d_ws, size_t ws_size,
                              hipStream_t stream) {
    const float*     X   = (const float*)d_in[0];
    const float*     L   = (const float*)d_in[1];
    const long long* tgt = (const long long*)d_in[2];

    const int n  = in_sizes[2];
    const int d  = in_sizes[0] / n;
    const int dl = in_sizes[1] / n;

    // workspace layout: small arrays first, then bf16 hi/lo split of X
    float*  sq     = (float*)d_ws;
    int*    pidx   = (int*)(sq + n);
    int*    nidx   = pidx + n;
    float*  loss_g = (float*)(nidx + n);
    float*  loss_l = loss_g + n;
    __bf16* Xhi    = (__bf16*)(loss_l + n);
    __bf16* Xlo    = Xhi + (size_t)n * d;

    const size_t need = (size_t)5 * n * sizeof(float)
                      + (size_t)2 * n * d * sizeof(__bf16);
    const bool fast = (ws_size >= need) && (d % 32 == 0) && (d <= DMAX)
                   && (n % 32 == 0);

    if (fast) {
        split_kernel<<<n, 256, 0, stream>>>(X, Xhi, Xlo, sq, d);
        mine_kernel_pre<<<n / 16, 256, 0, stream>>>(Xhi, Xlo, sq, tgt,
                                                    pidx, nidx, n, d);
    } else {
        sq_kernel<<<n, 256, 0, stream>>>(X, sq, d);
        mine_kernel_fly<<<n / 16, 256, 0, stream>>>(X, sq, tgt, pidx, nidx, n, d);
    }
    local_kernel<<<n / 8, 256, 0, stream>>>(X, L, sq, pidx, nidx,
                                            loss_g, loss_l, n, d, dl);
    reduce_kernel<<<1, 256, 0, stream>>>(loss_g, loss_l, (float*)d_out, n);
}
